// GNN_79637283602863
// MI455X (gfx1250) — matrix operand order repaired and checked by reading, not compiled, not run on hardware
//
#include <hip/hip_runtime.h>
#include <hip/hip_bf16.h>

// ---------------------------------------------------------------------------
// GATv2 GNN forward (4 layers) for MI455X / gfx1250.
// Dense node transforms run on V_WMMA_F32_16X16X32_BF16 (wave32 WMMA),
// 32x64 output tile per wave, transposed bf16 weights so both A and B
// fragments load as contiguous global_load_b128 pairs.
// Edge softmax/aggregation are handled with wave-per-edge kernels + atomics.
// ---------------------------------------------------------------------------

typedef __attribute__((ext_vector_type(16))) __bf16 v16bf;
typedef __attribute__((ext_vector_type(8)))  __bf16 v8bf;
typedef __attribute__((ext_vector_type(8)))  float  v8f;

#define N_NODES 8192
#define N_EDGES 65536
#define EP      (N_EDGES + N_NODES)   // edges including self loops = 73728
#define N_GRAPH 64
#define EAD     21                    // encoded edge-attr dim
#define OUTD    512                   // final feature dim

// order-preserving float -> uint mapping for atomicMax-based segment max
static __device__ __forceinline__ unsigned fmap_ord(float f) {
  unsigned u = __float_as_uint(f);
  return (u & 0x80000000u) ? ~u : (u | 0x80000000u);
}
static __device__ __forceinline__ float funmap_ord(unsigned u) {
  return (u & 0x80000000u) ? __uint_as_float(u & 0x7FFFFFFFu)
                           : __uint_as_float(~u);
}

// ------------------------------- utility -----------------------------------
__global__ void k_fill_f32(float* p, long n, float v) {
  long i = (long)blockIdx.x * blockDim.x + threadIdx.x;
  if (i < n) p[i] = v;
}
__global__ void k_fill_u32(unsigned* p, long n, unsigned v) {
  long i = (long)blockIdx.x * blockDim.x + threadIdx.x;
  if (i < n) p[i] = v;
}

// --------------------------- feature encoding ------------------------------
// h0[n, 0:74] = [atom_emb(64) | x[:,1:9] | bool_emb(2)]
__global__ void k_encode_nodes(const float* __restrict__ x,
                               const float* __restrict__ atom_emb,
                               const float* __restrict__ bool_emb,
                               float* __restrict__ h) {
  long i = (long)blockIdx.x * blockDim.x + threadIdx.x;
  if (i >= (long)N_NODES * 74) return;
  int n = (int)(i / 74), c = (int)(i % 74);
  const float* xr = x + (long)n * 10;
  float v;
  if (c < 64)      v = atom_emb[(int)xr[0] * 64 + c];
  else if (c < 72) v = xr[(c - 64) + 1];
  else             v = bool_emb[(int)xr[9] * 2 + (c - 72)];
  h[(long)n * 74 + c] = v;
}

// ea[e, 0:21] = [bond_emb(16) | edge_attr[:,1] | bool_emb(2) | bool_emb(2)]
__global__ void k_encode_edges(const float* __restrict__ edge_attr,
                               const float* __restrict__ bond_emb,
                               const float* __restrict__ bool_emb,
                               float* __restrict__ ea) {
  long i = (long)blockIdx.x * blockDim.x + threadIdx.x;
  if (i >= (long)N_EDGES * EAD) return;
  int e = (int)(i / EAD), c = (int)(i % EAD);
  const float* ar = edge_attr + (long)e * 4;
  float v;
  if (c < 16)       v = bond_emb[(int)ar[0] * 16 + c];
  else if (c == 16) v = ar[1];
  else if (c < 19)  v = bool_emb[(int)ar[2] * 2 + (c - 17)];
  else              v = bool_emb[(int)ar[3] * 2 + (c - 19)];
  ea[(long)e * EAD + c] = v;
}

// self-loop attr = mean of incoming edge attrs (fill_value='mean')
__global__ void k_loop_acc(const float* __restrict__ ea,
                           const int* __restrict__ eidx,
                           float* __restrict__ acc, float* __restrict__ cnt) {
  int e = blockIdx.x * blockDim.x + threadIdx.x;
  if (e >= N_EDGES) return;
  int t = eidx[N_EDGES + e];
  atomicAdd(&cnt[t], 1.0f);
  for (int k = 0; k < EAD; ++k)
    atomicAdd(&acc[(long)t * EAD + k], ea[(long)e * EAD + k]);
}
__global__ void k_loop_fin(const float* __restrict__ acc,
                           const float* __restrict__ cnt,
                           float* __restrict__ ea) {
  long i = (long)blockIdx.x * blockDim.x + threadIdx.x;
  if (i >= (long)N_NODES * EAD) return;
  int n = (int)(i / EAD);
  ea[((long)N_EDGES + n) * EAD + (i % EAD)] = acc[i] / fmaxf(cnt[n], 1.0f);
}

// ------------------------- bf16 staging (K padding) -------------------------
// activations: f32 [R, Cin] -> bf16 [R, Cp] zero-padded cols (K contiguous)
__global__ void k_pad_cols_bf16(const float* __restrict__ src,
                                __bf16* __restrict__ dst,
                                int R, int Cin, int Cp) {
  long i = (long)blockIdx.x * blockDim.x + threadIdx.x;
  if (i >= (long)R * Cp) return;
  int r = (int)(i / Cp), c = (int)(i % Cp);
  dst[i] = (c < Cin) ? (__bf16)src[(long)r * Cin + c] : (__bf16)0.0f;
}
// weights: f32 [Din, DO] -> bf16 TRANSPOSED [DO, Kp], zero-padded K.
// K-contiguous layout makes B fragments contiguous 32B runs (b128 loads).
__global__ void k_w_transpose_bf16(const float* __restrict__ W,
                                   __bf16* __restrict__ Wt,
                                   int Din, int Kp, int DO) {
  long i = (long)blockIdx.x * blockDim.x + threadIdx.x;
  if (i >= (long)DO * Kp) return;
  int c = (int)(i / Kp), k = (int)(i % Kp);
  Wt[i] = (k < Din) ? (__bf16)W[(long)k * DO + c] : (__bf16)0.0f;
}

// ------------------------------ WMMA GEMM ----------------------------------
// C[M,N] = A[M,K](bf16, row-major) x Wt[N,K](bf16, K-contiguous) + bias.
// One wave -> 32x64 tile: 2 A fragments x 4 B fragments -> 8 WMMAs per K-step.
__global__ void __launch_bounds__(256)
k_wmma_gemm_bias(const __bf16* __restrict__ A, const __bf16* __restrict__ Bt,
                 const float* __restrict__ bias, float* __restrict__ C,
                 int M, int K, int Ncols) {
  const int lane  = threadIdx.x & 31;
  const int wave  = (int)(((long)blockIdx.x * blockDim.x + threadIdx.x) >> 5);
  const int tilesN = Ncols >> 6;
  const int totalTiles = (M >> 5) * tilesN;
  if (wave >= totalTiles) return;            // lane-uniform: EXEC stays full
  const int tm   = wave / tilesN;
  const int tn   = wave - tm * tilesN;
  const int half = lane >> 4;                // lane half selects K sub-chunk
  const int l15  = lane & 15;
  const long rowA0 = (long)tm * 32 + l15;    // A rows: M = lane%16 (both halves)
  const long rowA1 = rowA0 + 16;
  const int n0   = tn * 64;

  v8f acc[2][4] = {};

  const __bf16* ap0 = A + rowA0 * K + 8 * half;
  const __bf16* ap1 = A + rowA1 * K + 8 * half;

  for (int kk = 0; kk < K; kk += 32) {
    // A fragments (16-bit A 16x32): lane half h holds K = kk+8h..+7, kk+16+8h..+7
    v8bf a0lo = *(const v8bf*)(ap0 + kk);
    v8bf a0hi = *(const v8bf*)(ap0 + kk + 16);
    v8bf a1lo = *(const v8bf*)(ap1 + kk);
    v8bf a1hi = *(const v8bf*)(ap1 + kk + 16);
    v16bf a0, a1;
#pragma unroll
    for (int i = 0; i < 8; ++i) {
      a0[i] = a0lo[i]; a0[8 + i] = a0hi[i];
      a1[i] = a1lo[i]; a1[8 + i] = a1hi[i];
    }
    // prefetch next K-step of A rows (emits global_prefetch_b8)
    __builtin_prefetch((const void*)(ap0 + kk + 32), 0, 1);
    __builtin_prefetch((const void*)(ap1 + kk + 32), 0, 1);

#pragma unroll
    for (int sub = 0; sub < 4; ++sub) {
      // B fragment (32x16): col = lane%16 (+16*sub), K rows kk+8*half..+7,
      // kk+16+8*half..+7 in the transposed weight layout.
      const __bf16* bp =
          Bt + (long)(n0 + sub * 16 + l15) * K + kk + 8 * half;
      v8bf blo = *(const v8bf*)(bp);
      v8bf bhi = *(const v8bf*)(bp + 16);
      v16bf b;
#pragma unroll
      for (int i = 0; i < 8; ++i) { b[i] = blo[i]; b[8 + i] = bhi[i]; }
      acc[0][sub] = __builtin_amdgcn_wmma_f32_16x16x32_bf16(
          false, a0, false, b, (short)0, acc[0][sub], false, false);
      acc[1][sub] = __builtin_amdgcn_wmma_f32_16x16x32_bf16(
          false, a1, false, b, (short)0, acc[1][sub], false, false);
    }
  }

  // D layout: lane col N = lane%16; VGPR v -> row M = v + 8*(lane/16)
#pragma unroll
  for (int mh = 0; mh < 2; ++mh) {
#pragma unroll
    for (int sub = 0; sub < 4; ++sub) {
      int col = n0 + sub * 16 + l15;
      float bv = bias[col];
#pragma unroll
      for (int v = 0; v < 8; ++v) {
        long r = (long)tm * 32 + mh * 16 + v + 8 * half;
        C[r * Ncols + col] = acc[mh][sub][v] + bv;
      }
    }
  }
}

// ------------------------------ edge kernels -------------------------------
// One wave per edge: logit a[e] = sum_c leaky(xl[s,c]+xr[t,c]+ (ea_e . We[:,c])) * att[c]
__global__ void __launch_bounds__(256)
k_edge_logits(const float* __restrict__ xl, const float* __restrict__ xr,
              const float* __restrict__ ea, const float* __restrict__ We,
              const float* __restrict__ att, const int* __restrict__ eidx,
              float* __restrict__ alog, unsigned* __restrict__ amax, int dout) {
  const int lane = threadIdx.x & 31;
  const int e = (int)(((long)blockIdx.x * blockDim.x + threadIdx.x) >> 5);
  if (e >= EP) return;
  int s, t;
  if (e < N_EDGES) { s = eidx[e]; t = eidx[N_EDGES + e]; }
  else             { s = e - N_EDGES; t = s; }          // self loop
  float w[EAD];
  const float* eav = ea + (long)e * EAD;                // lane-uniform -> scalar
#pragma unroll
  for (int k = 0; k < EAD; ++k) w[k] = eav[k];
  float acc = 0.f;
  for (int c = lane; c < dout; c += 32) {
    float m = xl[(long)s * dout + c] + xr[(long)t * dout + c];
#pragma unroll
    for (int k = 0; k < EAD; ++k) m += w[k] * We[(long)k * dout + c];
    float lr = (m > 0.f) ? m : 0.2f * m;                // leaky_relu(0.2)
    acc += lr * att[c];
  }
#pragma unroll
  for (int off = 16; off > 0; off >>= 1) acc += __shfl_xor(acc, off, 32);
  if (lane == 0) {
    alog[e] = acc;
    atomicMax(&amax[t], fmap_ord(acc));                 // segment max
  }
}

__global__ void k_edge_exp(const float* __restrict__ alog,
                           const unsigned* __restrict__ amax,
                           const int* __restrict__ eidx,
                           float* __restrict__ aex, float* __restrict__ denom) {
  int e = blockIdx.x * blockDim.x + threadIdx.x;
  if (e >= EP) return;
  int t = (e < N_EDGES) ? eidx[N_EDGES + e] : (e - N_EDGES);
  float ex = __expf(alog[e] - funmap_ord(amax[t]));
  aex[e] = ex;
  atomicAdd(&denom[t], ex);
}

// out[t, :] += (ex/denom[t]) * xl[s, :]
__global__ void __launch_bounds__(256)
k_edge_aggregate(const float* __restrict__ xl, const float* __restrict__ aex,
                 const float* __restrict__ denom, const int* __restrict__ eidx,
                 float* __restrict__ out, int dout) {
  const int lane = threadIdx.x & 31;
  const int e = (int)(((long)blockIdx.x * blockDim.x + threadIdx.x) >> 5);
  if (e >= EP) return;
  int s, t;
  if (e < N_EDGES) { s = eidx[e]; t = eidx[N_EDGES + e]; }
  else             { s = e - N_EDGES; t = s; }
  float alpha = aex[e] / denom[t];
  for (int c = lane; c < dout; c += 32)
    atomicAdd(&out[(long)t * dout + c], alpha * xl[(long)s * dout + c]);
}

__global__ void k_bias_relu(float* __restrict__ h, const float* __restrict__ b,
                            int dout, long total) {
  long i = (long)blockIdx.x * blockDim.x + threadIdx.x;
  if (i >= total) return;
  h[i] = fmaxf(h[i] + b[(int)(i % dout)], 0.f);
}

// ------------------------------ graph pooling ------------------------------
__global__ void __launch_bounds__(256)
k_pool_sum(const float* __restrict__ h, const int* __restrict__ batch,
           float* __restrict__ out, float* __restrict__ gcnt) {
  const int lane = threadIdx.x & 31;
  const int n = (int)(((long)blockIdx.x * blockDim.x + threadIdx.x) >> 5);
  if (n >= N_NODES) return;
  int g = batch[n];
  if (lane == 0) atomicAdd(&gcnt[g], 1.0f);
  for (int c = lane; c < OUTD; c += 32)
    atomicAdd(&out[(long)g * OUTD + c], h[(long)n * OUTD + c]);
}
__global__ void k_pool_div(float* __restrict__ out, const float* __restrict__ gcnt) {
  int i = blockIdx.x * blockDim.x + threadIdx.x;
  if (i >= N_GRAPH * OUTD) return;
  out[i] /= fmaxf(gcnt[i / OUTD], 1.0f);
}

// ------------------------------- launcher ----------------------------------
extern "C" void kernel_launch(void* const* d_in, const int* in_sizes, int n_in,
                              void* d_out, int out_size, void* d_ws, size_t ws_size,
                              hipStream_t stream) {
  (void)in_sizes; (void)n_in; (void)out_size; (void)ws_size;
  const float* x         = (const float*)d_in[0];
  const float* edge_attr = (const float*)d_in[1];
  const float* atom_emb  = (const float*)d_in[2];
  const float* bond_emb  = (const float*)d_in[3];
  const float* bool_emb  = (const float*)d_in[4];
  const int*   eidx      = (const int*)d_in[5];
  const int*   batch     = (const int*)d_in[6];
  const float *Wl[4], *bl[4], *Wr[4], *br[4], *We[4], *att[4], *bb[4];
  for (int l = 0; l < 4; ++l) {
    const int base = 7 + 7 * l;
    Wl[l]  = (const float*)d_in[base + 0];
    bl[l]  = (const float*)d_in[base + 1];
    Wr[l]  = (const float*)d_in[base + 2];
    br[l]  = (const float*)d_in[base + 3];
    We[l]  = (const float*)d_in[base + 4];
    att[l] = (const float*)d_in[base + 5];
    bb[l]  = (const float*)d_in[base + 6];
  }

  // carve workspace (256B aligned blocks)
  char* p = (char*)d_ws;
  auto carve = [&](size_t bytes) -> void* {
    char* r = p; p += (bytes + 255) & ~(size_t)255; return (void*)r;
  };
  float*    hA    = (float*)carve((size_t)N_NODES * 2048 * 4);
  float*    hB    = (float*)carve((size_t)N_NODES * 2048 * 4);
  float*    xl    = (float*)carve((size_t)N_NODES * 2048 * 4);
  float*    xr    = (float*)carve((size_t)N_NODES * 2048 * 4);
  __bf16*   Abf   = (__bf16*)carve((size_t)N_NODES * 2048 * 2);
  __bf16*   Wbf   = (__bf16*)carve((size_t)2048 * 2048 * 2);
  float*    ea    = (float*)carve((size_t)EP * EAD * 4);
  float*    lacc  = (float*)carve((size_t)N_NODES * EAD * 4);
  float*    lcnt  = (float*)carve((size_t)N_NODES * 4);
  float*    alog  = (float*)carve((size_t)EP * 4);
  float*    aex   = (float*)carve((size_t)EP * 4);
  unsigned* amax  = (unsigned*)carve((size_t)N_NODES * 4);
  float*    denom = (float*)carve((size_t)N_NODES * 4);
  float*    gcnt  = (float*)carve((size_t)N_GRAPH * 4);

  const int B = 256;   // 8 wave32 per workgroup
  auto nb = [&](long n) { return (unsigned)((n + B - 1) / B); };

  // feature + edge encoding, self-loop mean attrs
  k_encode_nodes<<<nb((long)N_NODES * 74), B, 0, stream>>>(x, atom_emb, bool_emb, hA);
  k_encode_edges<<<nb((long)N_EDGES * EAD), B, 0, stream>>>(edge_attr, bond_emb, bool_emb, ea);
  k_fill_f32<<<nb((long)N_NODES * EAD), B, 0, stream>>>(lacc, (long)N_NODES * EAD, 0.f);
  k_fill_f32<<<nb(N_NODES), B, 0, stream>>>(lcnt, N_NODES, 0.f);
  k_loop_acc<<<nb(N_EDGES), B, 0, stream>>>(ea, eidx, lacc, lcnt);
  k_loop_fin<<<nb((long)N_NODES * EAD), B, 0, stream>>>(lacc, lcnt, ea);

  const int din[4] = {74, 2048, 1024, 512};
  const int kp[4]  = {96, 2048, 1024, 512};   // K padded to multiple of 32
  const int dov[4] = {2048, 1024, 512, 512};

  float* h  = hA;
  float* hn = hB;
  for (int l = 0; l < 4; ++l) {
    const int DI = din[l], KP = kp[l], DO = dov[l];
    const long waves = (long)(N_NODES / 32) * (DO / 64);

    k_pad_cols_bf16<<<nb((long)N_NODES * KP), B, 0, stream>>>(h, Abf, N_NODES, DI, KP);
    k_w_transpose_bf16<<<nb((long)DO * KP), B, 0, stream>>>(Wl[l], Wbf, DI, KP, DO);
    k_wmma_gemm_bias<<<nb(waves * 32), B, 0, stream>>>(Abf, Wbf, bl[l], xl, N_NODES, KP, DO);
    k_w_transpose_bf16<<<nb((long)DO * KP), B, 0, stream>>>(Wr[l], Wbf, DI, KP, DO);
    k_wmma_gemm_bias<<<nb(waves * 32), B, 0, stream>>>(Abf, Wbf, br[l], xr, N_NODES, KP, DO);

    k_fill_u32<<<nb(N_NODES), B, 0, stream>>>(amax, N_NODES, 0u);
    k_fill_f32<<<nb(N_NODES), B, 0, stream>>>(denom, N_NODES, 0.f);
    k_edge_logits<<<nb((long)EP * 32), B, 0, stream>>>(xl, xr, ea, We[l], att[l],
                                                       eidx, alog, amax, DO);
    k_edge_exp<<<nb(EP), B, 0, stream>>>(alog, amax, eidx, aex, denom);
    k_fill_f32<<<nb((long)N_NODES * DO), B, 0, stream>>>(hn, (long)N_NODES * DO, 0.f);
    k_edge_aggregate<<<nb((long)EP * 32), B, 0, stream>>>(xl, aex, denom, eidx, hn, DO);
    k_bias_relu<<<nb((long)N_NODES * DO), B, 0, stream>>>(hn, bb[l], DO, (long)N_NODES * DO);

    float* tmp = h; h = hn; hn = tmp;
  }

  // mean pooling per graph into d_out [64, 512]
  k_fill_f32<<<nb((long)N_GRAPH * OUTD), B, 0, stream>>>((float*)d_out, (long)N_GRAPH * OUTD, 0.f);
  k_fill_f32<<<nb(N_GRAPH), B, 0, stream>>>(gcnt, N_GRAPH, 0.f);
  k_pool_sum<<<nb((long)N_NODES * 32), B, 0, stream>>>(h, batch, (float*)d_out, gcnt);
  k_pool_div<<<nb((long)N_GRAPH * OUTD), B, 0, stream>>>((float*)d_out, gcnt);
}
